// MultiHeadAttention_16896401343252
// MI455X (gfx1250) — matrix operand order repaired and checked by reading, not compiled, not run on hardware
//
#include <hip/hip_runtime.h>
#include <hip/hip_bf16.h>

typedef __attribute__((ext_vector_type(16))) __bf16 bf16x16;
typedef __attribute__((ext_vector_type(8)))  __bf16 bf16x8;
typedef __attribute__((ext_vector_type(8)))  float  floatx8;
typedef __attribute__((ext_vector_type(4)))  unsigned int uint32x4;
typedef __attribute__((ext_vector_type(8)))  int int32x8;
typedef __attribute__((ext_vector_type(4)))  int int32x4;

#define WMMA_BF16(a, b, c) \
  __builtin_amdgcn_wmma_f32_16x16x32_bf16(false, (a), false, (b), (short)0, (c), false, false)

// Tensor Data Mover builtin: 5-arg on ROCm clang-22, 6-arg on therock clang-23.
#if __has_include(<hip/amd_detail/amd_gfx1250_TDM.h>)
#define TDM_LOAD_TO_LDS(g0, g1, g2, g3) \
  __builtin_amdgcn_tensor_load_to_lds((g0), (g1), (g2), (g3), (int32x8){0, 0, 0, 0, 0, 0, 0, 0}, 0)
#else
#define TDM_LOAD_TO_LDS(g0, g1, g2, g3) \
  __builtin_amdgcn_tensor_load_to_lds((g0), (g1), (g2), (g3), 0)
#endif

constexpr int Bc = 4, Sc = 2048, Dc = 1024, Hc = 16, DHc = 64;

// ---------------- fragment loaders (CDNA5 ISA 7.12.2 layouts, wide LDS reads) ----------------

// A-matrix 16x32 bf16, element (m,k) at base[m*ld + k].
// Lane (m = lane&15, half = lane>>4) needs k in [8h, 8h+8) and [16+8h, 16+8h+8) -> two b128 loads.
__device__ inline bf16x16 load_frag_a(const __bf16* base, int ld) {
  const int lane = threadIdx.x & 31;
  const int h8 = (lane >> 4) << 3;  // 0 or 8
  const __bf16* r = base + (lane & 15) * ld + h8;
  union { bf16x16 v; bf16x8 h[2]; } u;
  u.h[0] = *(const bf16x8*)(r);
  u.h[1] = *(const bf16x8*)(r + 16);
  return u.v;
}

// B-matrix 32x16 bf16 stored transposed: element (k,n) at base[n*ld + k].
// Lane (n = lane&15, half) needs k in [8h, 8h+8) and [16+8h, 16+8h+8) -> two b128 loads.
__device__ inline bf16x16 load_frag_bt(const __bf16* base, int ld) {
  const int lane = threadIdx.x & 31;
  const int h16 = (lane >> 4) << 3;  // 0 or 8
  const __bf16* r = base + (lane & 15) * ld + h16;
  union { bf16x16 v; bf16x8 h[2]; } u;
  u.h[0] = *(const bf16x8*)(r);
  u.h[1] = *(const bf16x8*)(r + 16);
  return u.v;
}

__device__ inline void store_pair_bf16(__bf16* dst, float a, float b) {
  union { __bf16 h[2]; unsigned u; } p;
  p.h[0] = (__bf16)a;
  p.h[1] = (__bf16)b;
  *(unsigned*)dst = p.u;  // dst is 4B-aligned by construction
}

// ---------------- kernel 1: qkv = x @ w_qkv + b, scatter to (B,H,S,DH) bf16 ----------------

__global__ __launch_bounds__(256) void qkv_gemm_kernel(
    const float* __restrict__ x, const float* __restrict__ w, const float* __restrict__ bias,
    __bf16* __restrict__ Qb, __bf16* __restrict__ Kb, __bf16* __restrict__ Vb) {
  constexpr int BM = 64, BN = 128, BK = 32;
  constexpr int LDA = 40, LDB = 40;  // padded: 80B rows, 16B-aligned, conflict-free b128
  constexpr int Ncols = 3 * Dc;      // 3072
  constexpr int Kdim = Dc;           // 1024
  __shared__ __align__(16) __bf16 As[BM * LDA];   // A row-major (m,k)
  __shared__ __align__(16) __bf16 Bs[BN * LDB];   // B transposed (n,k)
  const int t = threadIdx.x;
  const int wid = t >> 5;
  const int wm = wid >> 2, wn = wid & 3;
  const int m0 = blockIdx.x * BM;
  const int n0 = blockIdx.y * BN;

  floatx8 acc[2][2] = {};

  for (int k0 = 0; k0 < Kdim; k0 += BK) {
#pragma unroll
    for (int i = 0; i < 4; ++i) {  // A tile 64x32: f32 pairs -> packed bf16x2
      int p = t + 256 * i;
      int r = p >> 4, cp = (p & 15) * 2;
      float2 xv = *(const float2*)(x + (size_t)(m0 + r) * Kdim + k0 + cp);
      store_pair_bf16(&As[r * LDA + cp], xv.x, xv.y);
    }
#pragma unroll
    for (int i = 0; i < 8; ++i) {  // B tile 32x128 -> transposed (n,k) packed pairs
      int p = t + 256 * i;
      int n = p & 127, kp = (p >> 7) * 2;
      float w0 = w[(size_t)(k0 + kp) * Ncols + n0 + n];
      float w1 = w[(size_t)(k0 + kp + 1) * Ncols + n0 + n];
      store_pair_bf16(&Bs[n * LDB + kp], w0, w1);
    }
    if (k0 + BK < Kdim) {  // stream next tiles toward L2 (global_prefetch_b8)
      __builtin_prefetch(x + (size_t)(m0 + (t >> 2)) * Kdim + k0 + BK + (t & 3) * 8, 0, 1);
      __builtin_prefetch(w + (size_t)(k0 + BK + (t >> 4)) * Ncols + n0 + (t & 15) * 8, 0, 1);
    }
    __syncthreads();
    bf16x16 a0 = load_frag_a(As + (wm * 32 + 0) * LDA, LDA);
    bf16x16 a1 = load_frag_a(As + (wm * 32 + 16) * LDA, LDA);
    bf16x16 b0 = load_frag_bt(Bs + (wn * 32 + 0) * LDB, LDB);
    bf16x16 b1 = load_frag_bt(Bs + (wn * 32 + 16) * LDB, LDB);
    acc[0][0] = WMMA_BF16(a0, b0, acc[0][0]);
    acc[0][1] = WMMA_BF16(a0, b1, acc[0][1]);
    acc[1][0] = WMMA_BF16(a1, b0, acc[1][0]);
    acc[1][1] = WMMA_BF16(a1, b1, acc[1][1]);
    __syncthreads();
  }

  // Epilogue: BN=128 divides D=1024, so the whole block lies in exactly one of Q/K/V.
  const int which = n0 >> 10;               // uniform: 0=Q,1=K,2=V
  __bf16* const dstbuf = (which == 0) ? Qb : ((which == 1) ? Kb : Vb);
  const float scale = (which == 0) ? 0.125f : 1.0f;  // fold 1/sqrt(DH) into Q
  const int dbase = n0 & 1023;
  const int lane = t & 31;
  const int hb = lane >> 4, nn = lane & 15;
#pragma unroll
  for (int tm = 0; tm < 2; ++tm)
#pragma unroll
    for (int tn = 0; tn < 2; ++tn) {
      const int d = dbase + wn * 32 + tn * 16 + nn;
      const int h = d >> 6, dh = d & 63;
      const float bv = bias[n0 + wn * 32 + tn * 16 + nn];
#pragma unroll
      for (int j = 0; j < 8; ++j) {
        int m = m0 + wm * 32 + tm * 16 + j + 8 * hb;
        int b = m >> 11, s = m & 2047;
        dstbuf[(((size_t)(b * Hc + h)) * Sc + s) * DHc + dh] =
            (__bf16)((acc[tm][tn][j] + bv) * scale);
      }
    }
}

// ---------------- kernel 2: flash attention, causal, bf16 WMMA + TDM K-staging ----------------

__global__ __launch_bounds__(128) void attn_kernel(
    const __bf16* __restrict__ Qb, const __bf16* __restrict__ Kb,
    const __bf16* __restrict__ Vb, __bf16* __restrict__ Ob) {
  constexpr int LDK = 72;  // 144B rows: 16B-aligned, spreads lanes over all bank groups
  constexpr int LDV = 40;  // V stored transposed (dh, key)
  constexpr int LDP = 40;
  __shared__ __align__(16) __bf16 Ks[32 * LDK];
  __shared__ __align__(16) __bf16 Vst[64 * LDV];
  __shared__ __align__(16) __bf16 Ps[4 * 16 * LDP];
  const int t = threadIdx.x;
  const int wid = t >> 5;
  const int lane = t & 31;
  const int hb = lane >> 4, nn = lane & 15;
  const int bh = blockIdx.x;    // b*H + h
  const int qblk = blockIdx.y;  // block of 64 query rows
  const int qrow = qblk * 64 + wid * 16;
  const size_t base = (size_t)bh * Sc * DHc;
  const int b = bh >> 4;
  const int h = bh & 15;
  __bf16* const Pw = Ps + wid * 16 * LDP;

  // Per-wave TDM slice: wave `wid` DMAs keys [8*wid, 8*wid+8) of each 32-key block.
  // D# group1 (invariant): data_size=2B, pad_enable, pad every 32 DWORDs (one 128B row)
  // by 4 DWORDs (-> LDK=72 elems), tensor 64x8, tile 64x8, row stride 64 elems.
  const unsigned lds_k_slice =
      (unsigned)(size_t)(Ks) + (unsigned)(wid * 8 * LDK * 2);
  const int32x8 g1 = {
      (int)((1u << 16) | (1u << 20) | (4u << 22) | (3u << 25)),  // dsize|pad_en|intv|amt
      (int)(64u << 16),  // tensor_dim0 = 64 (low 16 into bits 63:48)
      (int)(8u << 16),   // tensor_dim0 hi=0, tensor_dim1 = 8
      (int)(64u << 16),  // tensor_dim1 hi=0, tile_dim0 = 64
      (int)(8u),         // tile_dim1 = 8, tile_dim2 = 0
      (int)(64u),        // tensor_dim0_stride = 64 (low 32)
      0, 0};             // stride hi, tensor_dim1_stride = 0
  const int32x4 gz = {0, 0, 0, 0};

  bf16x16 qf0 = load_frag_a(Qb + base + (size_t)qrow * DHc, DHc);       // dh 0..31
  bf16x16 qf1 = load_frag_a(Qb + base + (size_t)qrow * DHc + 32, DHc);  // dh 32..63

  floatx8 o[4] = {};
  float ms[8], ls[8];
#pragma unroll
  for (int j = 0; j < 8; ++j) { ms[j] = -1e30f; ls[j] = 0.f; }

  const int kend = qblk * 64 + 64;
  for (int kb = 0; kb < kend; kb += 32) {
    const __bf16* gK = Kb + base + (size_t)kb * DHc;
    const __bf16* gV = Vb + base + (size_t)kb * DHc;

    {  // K: 32x64 bf16 block via Tensor Data Mover (async DMA, HW row padding)
      const unsigned long long ga = (unsigned long long)(size_t)(gK + wid * 8 * DHc);
      const uint32x4 g0 = {1u,  // count=1, user descriptor, no gather
                           lds_k_slice, (unsigned)ga,
                           (unsigned)(ga >> 32) | (2u << 30)};  // addr hi | type=2
      TDM_LOAD_TO_LDS(g0, g1, gz, gz);
    }
#pragma unroll
    for (int i = 0; i < 16; ++i) {  // V: transpose into (dh, key)
      int f = t + 128 * i;
      int key = f >> 6, dh = f & 63;
      Vst[dh * LDV + key] = gV[f];
    }
    if (kb + 32 < kend) {  // prefetch next V block (global_prefetch_b8)
      __builtin_prefetch(gV + 32 * DHc + t * 16, 0, 1);
    }
    __builtin_amdgcn_s_wait_tensorcnt((short)0);  // this wave's K slice landed
    __syncthreads();                              // publish all slices + V

    if (kb <= qrow + 15) {  // wave has at least one unmasked key in this block
      floatx8 c0 = {}, c1 = {};
      {
        bf16x16 bk0  = load_frag_bt(Ks, LDK);             // keys kb..kb+15, dh 0..31
        bf16x16 bk0h = load_frag_bt(Ks + 32, LDK);        // dh 32..63
        c0 = WMMA_BF16(qf0, bk0, c0);
        c0 = WMMA_BF16(qf1, bk0h, c0);
        bf16x16 bk1  = load_frag_bt(Ks + 16 * LDK, LDK);  // keys kb+16..kb+31
        bf16x16 bk1h = load_frag_bt(Ks + 16 * LDK + 32, LDK);
        c1 = WMMA_BF16(qf0, bk1, c1);
        c1 = WMMA_BF16(qf1, bk1h, c1);
      }
      float alpha[8];
#pragma unroll
      for (int j = 0; j < 8; ++j) {
        int row = qrow + j + 8 * hb;
        float s0 = c0[j], s1 = c1[j];
        if (kb + nn > row)      s0 = -1e30f;  // causal mask
        if (kb + 16 + nn > row) s1 = -1e30f;
        float mcur = fmaxf(s0, s1);
#pragma unroll
        for (int msk = 1; msk < 16; msk <<= 1)
          mcur = fmaxf(mcur, __shfl_xor(mcur, msk, 32));
        float mnew = fmaxf(ms[j], mcur);
        float al = __expf(ms[j] - mnew);
        float p0 = __expf(s0 - mnew);
        float p1 = __expf(s1 - mnew);
        float rs = p0 + p1;
#pragma unroll
        for (int msk = 1; msk < 16; msk <<= 1)
          rs += __shfl_xor(rs, msk, 32);
        ls[j] = ls[j] * al + rs;
        ms[j] = mnew;
        alpha[j] = al;
        __bf16* pw = Pw + (j + 8 * hb) * LDP;  // P row-major 16x32 (padded)
        pw[nn] = (__bf16)p0;
        pw[16 + nn] = (__bf16)p1;
      }
#pragma unroll
      for (int d2 = 0; d2 < 4; ++d2)  // rescale running O
#pragma unroll
        for (int j = 0; j < 8; ++j) o[d2][j] *= alpha[j];

      bf16x16 pa = load_frag_a(Pw, LDP);  // same-wave LDS RAW: DS is in-order
#pragma unroll
      for (int d2 = 0; d2 < 4; ++d2) {
        bf16x16 bv = load_frag_bt(Vst + (d2 * 16) * LDV, LDV);  // (k=key, n=dh)
        o[d2] = WMMA_BF16(pa, bv, o[d2]);
      }
    }
    __syncthreads();
  }

#pragma unroll
  for (int d2 = 0; d2 < 4; ++d2)
#pragma unroll
    for (int j = 0; j < 8; ++j) {
      int s = qrow + j + 8 * hb;
      float v = o[d2][j] / ls[j];
      Ob[((size_t)(b * Sc + s)) * Dc + h * DHc + d2 * 16 + nn] = (__bf16)v;
    }
}

// ---------------- kernel 3: out = attn_out @ w_out + b_out (f32 result) ----------------

__global__ __launch_bounds__(256) void out_gemm_kernel(
    const __bf16* __restrict__ A, const float* __restrict__ w,
    const float* __restrict__ bias, float* __restrict__ out) {
  constexpr int BM = 64, BN = 128, BK = 32;
  constexpr int LDA = 40, LDB = 40;
  constexpr int N = Dc, Kdim = Dc;
  __shared__ __align__(16) __bf16 As[BM * LDA];
  __shared__ __align__(16) __bf16 Bs[BN * LDB];
  const int t = threadIdx.x;
  const int wid = t >> 5;
  const int wm = wid >> 2, wn = wid & 3;
  const int m0 = blockIdx.x * BM;
  const int n0 = blockIdx.y * BN;
  floatx8 acc[2][2] = {};
  for (int k0 = 0; k0 < Kdim; k0 += BK) {
    {  // A tile 64x32 bf16: one b128 copy per thread
      int r = t >> 2, co = (t & 3) * 8;
      *(uint4*)(As + r * LDA + co) =
          *(const uint4*)(A + (size_t)(m0 + r) * Kdim + k0 + co);
    }
#pragma unroll
    for (int i = 0; i < 8; ++i) {  // B tile 32x128 -> transposed packed pairs
      int p = t + 256 * i;
      int n = p & 127, kp = (p >> 7) * 2;
      float w0 = w[(size_t)(k0 + kp) * N + n0 + n];
      float w1 = w[(size_t)(k0 + kp + 1) * N + n0 + n];
      store_pair_bf16(&Bs[n * LDB + kp], w0, w1);
    }
    if (k0 + BK < Kdim) {
      __builtin_prefetch(A + (size_t)(m0 + (t >> 2)) * Kdim + k0 + BK + (t & 3) * 8, 0, 1);
      __builtin_prefetch(w + (size_t)(k0 + BK + (t >> 4)) * N + n0 + (t & 15) * 8, 0, 1);
    }
    __syncthreads();
    bf16x16 a0 = load_frag_a(As + (wm * 32) * LDA, LDA);
    bf16x16 a1 = load_frag_a(As + (wm * 32 + 16) * LDA, LDA);
    bf16x16 b0 = load_frag_bt(Bs + (wn * 32) * LDB, LDB);
    bf16x16 b1 = load_frag_bt(Bs + (wn * 32 + 16) * LDB, LDB);
    acc[0][0] = WMMA_BF16(a0, b0, acc[0][0]);
    acc[0][1] = WMMA_BF16(a0, b1, acc[0][1]);
    acc[1][0] = WMMA_BF16(a1, b0, acc[1][0]);
    acc[1][1] = WMMA_BF16(a1, b1, acc[1][1]);
    __syncthreads();
  }
  const int lane = t & 31, hb = lane >> 4, nn = lane & 15;
#pragma unroll
  for (int tm = 0; tm < 2; ++tm)
#pragma unroll
    for (int tn = 0; tn < 2; ++tn) {
      const int n = n0 + wn * 32 + tn * 16 + nn;
      const float bv = bias[n];
#pragma unroll
      for (int j = 0; j < 8; ++j) {
        int m = m0 + wm * 32 + tm * 16 + j + 8 * hb;
        out[(size_t)m * N + n] = acc[tm][tn][j] + bv;
      }
    }
}

// ---------------- launch ----------------

extern "C" void kernel_launch(void* const* d_in, const int* in_sizes, int n_in,
                              void* d_out, int out_size, void* d_ws, size_t ws_size,
                              hipStream_t stream) {
  const float* x     = (const float*)d_in[0];
  const float* w_qkv = (const float*)d_in[1];
  const float* b_qkv = (const float*)d_in[2];
  const float* w_out = (const float*)d_in[3];
  const float* b_out = (const float*)d_in[4];
  // d_in[5] (mask) unused: causal mask computed analytically.

  const size_t elems = (size_t)Bc * Hc * Sc * DHc;  // 8M elements
  __bf16* Qb = (__bf16*)d_ws;
  __bf16* Kb = Qb + elems;
  __bf16* Vb = Kb + elems;
  __bf16* Ob = Vb + elems;  // attention output, (B,S,D) bf16

  dim3 g1((Bc * Sc) / 64, (3 * Dc) / 128);
  hipLaunchKernelGGL(qkv_gemm_kernel, g1, dim3(256), 0, stream, x, w_qkv, b_qkv, Qb, Kb, Vb);

  dim3 g2(Bc * Hc, Sc / 64);
  hipLaunchKernelGGL(attn_kernel, g2, dim3(128), 0, stream, Qb, Kb, Vb, Ob);

  dim3 g3((Bc * Sc) / 64, Dc / 128);
  hipLaunchKernelGGL(out_gemm_kernel, g3, dim3(256), 0, stream, Ob, w_out, b_out, (float*)d_out);
}
